// LocalInferenceModeling_4595615006871
// MI455X (gfx1250) — matrix operand order repaired and checked by reading, not compiled, not run on hardware
//
#include <hip/hip_runtime.h>
#include <hip/hip_bf16.h>
#include <math.h>

// Problem constants (match reference)
#define BB 32
#define LL 512
#define HH 1024

typedef __attribute__((ext_vector_type(16))) _Float16 v16h;
typedef __attribute__((ext_vector_type(4)))  _Float16 v4h;
typedef __attribute__((ext_vector_type(8)))  float    v8f;

// ---------------------------------------------------------------------------
// WMMA helpers (CDNA5 gfx1250, wave32).
// A-frag (16x32 f16): lane l (row = l&15): K = half*16 + e, e=0..15 contiguous
//                     (half = l>>4).
// B-frag (32x16 f16): lane l (col n = l&15): K = half*16 + e, e=0..15 contiguous.
// C/D (16x16 f32):    elem v: row m = 8*half + v, col n = l&15.
// ---------------------------------------------------------------------------
__device__ __forceinline__ v8f wmma16(v16h a, v16h b, v8f c) {
  return __builtin_amdgcn_wmma_f32_16x16x32_f16(
      /*neg_a=*/false, a, /*neg_b=*/false, b,
      /*c_mod=*/(short)0, c, /*reuse_a=*/false, /*reuse_b=*/false);
}

__device__ __forceinline__ v16h load_a_h(const _Float16* __restrict__ p) {
  v16h a;
#pragma unroll
  for (int e = 0; e < 8; ++e) a[e]     = p[e];
#pragma unroll
  for (int e = 0; e < 8; ++e) a[e + 8] = p[e + 8];
  return a;
}
__device__ __forceinline__ v16h load_b_h(const _Float16* __restrict__ p) {
  v16h b;
#pragma unroll
  for (int e = 0; e < 16; ++e) b[e] = p[e];
  return b;
}

// ---------------------------------------------------------------------------
// Kernel 0: elementwise f32 -> f16 copy (vectorized), one-time staging.
// ---------------------------------------------------------------------------
__global__ void cvt_f16(const float* __restrict__ x, _Float16* __restrict__ xh) {
  const size_t idx = ((size_t)blockIdx.x * blockDim.x + threadIdx.x) * 4;
  const float4 v = *(const float4*)(x + idx);
  v4h o;
  o[0] = (_Float16)v.x; o[1] = (_Float16)v.y;
  o[2] = (_Float16)v.z; o[3] = (_Float16)v.w;
  *(v4h*)(xh + idx) = o;
}

// ---------------------------------------------------------------------------
// Kernel 1: e[b,i,j] = sum_h x1[b,i,h] * x2[b,j,h].
// Each wave: 16(i) x 64(j) strip; the 8 waves of a block share j0 (identical
// B-frag addresses -> WGP$ hits) and differ in i0. Simple loop body: the
// scheduler interleaves next-tile load clauses with WMMAs without spilling.
// ---------------------------------------------------------------------------
__global__ void gemm_scores(const _Float16* __restrict__ x1h,
                            const _Float16* __restrict__ x2h,
                            float* __restrict__ e) {
  const int wave = (blockIdx.x * blockDim.x + threadIdx.x) >> 5;
  const int lane = threadIdx.x & 31;
  const int half = lane >> 4;
  const int lr   = lane & 15;

  const int per_batch = (LL / 16) * (LL / 64);  // 256 strips
  const int b  = wave / per_batch;
  const int r  = wave % per_batch;
  const int i0 = (r & 31) * 16;          // varies across waves of a block
  const int j0 = (r >> 5) * 64;          // shared by 32 consecutive waves

  const _Float16* __restrict__ Ab = x1h + (size_t)b * LL * HH;
  const _Float16* __restrict__ Bb = x2h + (size_t)b * LL * HH;
  const _Float16* __restrict__ arow = Ab + (size_t)(i0 + lr) * HH + half * 16;

  v8f acc[4] = {};
  for (int k0 = 0; k0 < HH; k0 += 32) {
    __builtin_prefetch(arow + k0 + 256, 0, 1);
    v16h af = load_a_h(arow + k0);
#pragma unroll
    for (int t = 0; t < 4; ++t) {
      const _Float16* bp = Bb + (size_t)(j0 + t * 16 + lr) * HH + k0 + half * 16;
      acc[t] = wmma16(af, load_b_h(bp), acc[t]);
    }
  }

  float* __restrict__ eb = e + (size_t)b * LL * LL;
#pragma unroll
  for (int t = 0; t < 4; ++t)
#pragma unroll
    for (int v = 0; v < 8; ++v)
      eb[(size_t)(i0 + 8 * half + v) * LL + (j0 + t * 16 + lr)] = acc[t][v];
}

// ---------------------------------------------------------------------------
// Kernel 2: row softmax over j with mask j >= len2  ->  a12[b,i,j] (f16)
// ---------------------------------------------------------------------------
__global__ void softmax_rows(const float* __restrict__ e,
                             const int* __restrict__ len2,
                             _Float16* __restrict__ a12) {
  const int b = blockIdx.x >> 9;          // / LL
  const int i = blockIdx.x & (LL - 1);
  const int n = len2[b];
  const int t = threadIdx.x;
  const float* __restrict__ row = e + ((size_t)b * LL + i) * LL;
  __shared__ float red[256];

  float m = -3.0e38f;
  if (t < n)       m = row[t];
  if (t + 256 < n) m = fmaxf(m, row[t + 256]);
  red[t] = m;
  __syncthreads();
#pragma unroll
  for (int s = 128; s > 0; s >>= 1) {
    if (t < s) red[t] = fmaxf(red[t], red[t + s]);
    __syncthreads();
  }
  m = red[0];
  __syncthreads();

  float p0 = (t < n)       ? __expf(row[t] - m)       : 0.0f;
  float p1 = (t + 256 < n) ? __expf(row[t + 256] - m) : 0.0f;
  red[t] = p0 + p1;
  __syncthreads();
#pragma unroll
  for (int s = 128; s > 0; s >>= 1) {
    if (t < s) red[t] += red[t + s];
    __syncthreads();
  }
  const float inv = 1.0f / red[0];

  _Float16* __restrict__ arow = a12 + ((size_t)b * LL + i) * LL;
  arow[t]       = (_Float16)(p0 * inv);
  arow[t + 256] = (_Float16)(p1 * inv);
}

// ---------------------------------------------------------------------------
// Kernel 3: column softmax over i with mask i >= len1 -> a21[b,j,i] (f16,
// row-major in i so the next GEMM's A-frag loads are contiguous).
// ---------------------------------------------------------------------------
__global__ void softmax_cols(const float* __restrict__ e,
                             const int* __restrict__ len1,
                             _Float16* __restrict__ a21) {
  const int b  = blockIdx.x / (LL / 32);
  const int j0 = (blockIdx.x % (LL / 32)) * 32;
  const int n  = len1[b];
  const int tx = threadIdx.x, ty = threadIdx.y;
  const float* __restrict__ eb = e + (size_t)b * LL * LL;

  __shared__ float red[8][32];
  __shared__ float cmax[32], csum[32];
  __shared__ float T[32][33];

  float m = -3.0e38f;
  for (int i = ty; i < n; i += 8) m = fmaxf(m, eb[(size_t)i * LL + j0 + tx]);
  red[ty][tx] = m;
  __syncthreads();
  if (ty == 0) {
    float mm = red[0][tx];
#pragma unroll
    for (int q = 1; q < 8; ++q) mm = fmaxf(mm, red[q][tx]);
    cmax[tx] = mm;
  }
  __syncthreads();
  m = cmax[tx];

  float s = 0.0f;
  for (int i = ty; i < n; i += 8) s += __expf(eb[(size_t)i * LL + j0 + tx] - m);
  red[ty][tx] = s;
  __syncthreads();
  if (ty == 0) {
    float ss = 0.0f;
#pragma unroll
    for (int q = 0; q < 8; ++q) ss += red[q][tx];
    csum[tx] = ss;
  }
  __syncthreads();
  const float inv = 1.0f / csum[tx];

  _Float16* __restrict__ ab = a21 + (size_t)b * LL * LL;
  for (int i0 = 0; i0 < LL; i0 += 32) {
#pragma unroll
    for (int q = 0; q < 4; ++q) {
      const int il = ty + 8 * q;
      const int i  = i0 + il;
      T[il][tx] = (i < n) ? __expf(eb[(size_t)i * LL + j0 + tx] - m) * inv : 0.0f;
    }
    __syncthreads();
#pragma unroll
    for (int q = 0; q < 4; ++q) {
      const int jj = ty + 8 * q;
      ab[(size_t)(j0 + jj) * LL + i0 + tx] = (_Float16)T[tx][jj];
    }
    __syncthreads();
  }
}

// ---------------------------------------------------------------------------
// Kernel 4: transpose+convert  x[B,L,H] f32  ->  xT[B,H,L] f16
// ---------------------------------------------------------------------------
__global__ void transpose_cvt(const float* __restrict__ x,
                              _Float16* __restrict__ xT) {
  const int tiles = (HH / 32) * (LL / 32);   // 512
  const int b  = blockIdx.x / tiles;
  const int r  = blockIdx.x % tiles;
  const int h0 = (r / (LL / 32)) * 32;
  const int l0 = (r % (LL / 32)) * 32;
  const int tx = threadIdx.x, ty = threadIdx.y;

  __shared__ float T[32][33];
  const float* __restrict__ xb = x + (size_t)b * LL * HH;
#pragma unroll
  for (int q = 0; q < 4; ++q) {
    const int l = l0 + ty + 8 * q;
    T[ty + 8 * q][tx] = xb[(size_t)l * HH + h0 + tx];
  }
  __syncthreads();
  _Float16* __restrict__ xTb = xT + (size_t)b * HH * LL;
#pragma unroll
  for (int q = 0; q < 4; ++q) {
    const int hh = ty + 8 * q;
    xTb[(size_t)(h0 + hh) * LL + l0 + tx] = (_Float16)T[tx][hh];
  }
}

// ---------------------------------------------------------------------------
// Kernel 5: x_tilde = attn @ V (via V^T), fused enhance epilogue:
// out[b,i,:] = [ u, xt, u - xt, u * xt ]   (out stride 4H)
// Waves of a block share h0 (B-frag reuse in WGP$); non-temporal stores for
// the 537 MB write-once output stream (keeps L2 for the working set).
// ---------------------------------------------------------------------------
__global__ void gemm_av_enhance(const _Float16* __restrict__ attn, // [B,L,L]
                                const _Float16* __restrict__ vT,   // [B,H,L]
                                const float* __restrict__ u,       // [B,L,H]
                                float* __restrict__ out) {         // [B,L,4H]
  const int wave = (blockIdx.x * blockDim.x + threadIdx.x) >> 5;
  const int lane = threadIdx.x & 31;
  const int half = lane >> 4;
  const int lr   = lane & 15;

  const int per_batch = (LL / 16) * (HH / 64);  // 512 strips
  const int b  = wave / per_batch;
  const int r  = wave % per_batch;
  const int i0 = (r & 31) * 16;          // varies across waves of a block
  const int h0 = (r >> 5) * 64;          // shared by 32 consecutive waves

  const _Float16* __restrict__ arow =
      attn + ((size_t)b * LL + i0 + lr) * LL + half * 16;
  const _Float16* __restrict__ vb = vT + (size_t)b * HH * LL;

  v8f acc[4] = {};
  for (int k0 = 0; k0 < LL; k0 += 32) {
    __builtin_prefetch(arow + k0 + 128, 0, 1);
    v16h af = load_a_h(arow + k0);
#pragma unroll
    for (int t = 0; t < 4; ++t) {
      const _Float16* bp = vb + (size_t)(h0 + t * 16 + lr) * LL + k0 + half * 16;
      acc[t] = wmma16(af, load_b_h(bp), acc[t]);
    }
  }

  const float* __restrict__ ub = u + (size_t)b * LL * HH;
  float* __restrict__ ob = out + (size_t)b * LL * (4 * HH);
#pragma unroll
  for (int t = 0; t < 4; ++t) {
#pragma unroll
    for (int v = 0; v < 8; ++v) {
      const int i = i0 + 8 * half + v;
      const int h = h0 + t * 16 + lr;
      const float xt = acc[t][v];
      const float xb = ub[(size_t)i * HH + h];
      float* orow = ob + (size_t)i * (4 * HH);
      __builtin_nontemporal_store(xb,      &orow[h]);
      __builtin_nontemporal_store(xt,      &orow[HH + h]);
      __builtin_nontemporal_store(xb - xt, &orow[2 * HH + h]);
      __builtin_nontemporal_store(xb * xt, &orow[3 * HH + h]);
    }
  }
}

// ---------------------------------------------------------------------------
extern "C" void kernel_launch(void* const* d_in, const int* in_sizes, int n_in,
                              void* d_out, int out_size, void* d_ws, size_t ws_size,
                              hipStream_t stream) {
  const float* x1   = (const float*)d_in[0];
  const int*   len1 = (const int*)d_in[1];
  const float* x2   = (const float*)d_in[2];
  const int*   len2 = (const int*)d_in[3];
  float* out = (float*)d_out;

  char* ws = (char*)d_ws;
  float*    e   = (float*)ws;                               // 32 MB
  _Float16* a12 = (_Float16*)(ws + (size_t)33554432);       // 16 MB
  _Float16* a21 = (_Float16*)(ws + (size_t)50331648);       // 16 MB
  _Float16* x1T = (_Float16*)(ws + (size_t)67108864);       // 32 MB
  _Float16* x2T = (_Float16*)(ws + (size_t)100663296);      // 32 MB
  _Float16* x1h = (_Float16*)(ws + (size_t)134217728);      // 32 MB
  _Float16* x2h = (_Float16*)(ws + (size_t)167772160);      // 32 MB

  const int nElem = BB * LL * HH;                           // 16,777,216

  // 0) one-time f16 staging (row-major + transposed)
  cvt_f16<<<dim3(nElem / 4 / 256), dim3(256), 0, stream>>>(x1, x1h);
  cvt_f16<<<dim3(nElem / 4 / 256), dim3(256), 0, stream>>>(x2, x2h);
  transpose_cvt<<<dim3(BB * 512), dim3(32, 8), 0, stream>>>(x1, x1T);
  transpose_cvt<<<dim3(BB * 512), dim3(32, 8), 0, stream>>>(x2, x2T);

  // 1) scores e = x1 @ x2^T : 8192 waves
  gemm_scores<<<dim3(1024), dim3(256), 0, stream>>>(x1h, x2h, e);

  // 2) softmaxes
  softmax_rows<<<dim3(BB * LL), dim3(256), 0, stream>>>(e, len2, a12);
  softmax_cols<<<dim3(BB * (LL / 32)), dim3(32, 8), 0, stream>>>(e, len1, a21);

  // 3) x1_tilde = a12 @ x2, enhance -> out1 ; x2_tilde = a21 @ x1 -> out2
  gemm_av_enhance<<<dim3(2048), dim3(256), 0, stream>>>(a12, x2T, x1, out);
  gemm_av_enhance<<<dim3(2048), dim3(256), 0, stream>>>(
      a21, x1T, x2, out + (size_t)BB * LL * 4 * HH);
}
